// MultiHeadAttention_60309930770550
// MI455X (gfx1250) — compile-verified, matrix operand order checked by reading, not run on hardware
//
#include <hip/hip_runtime.h>
#include <hip/hip_bf16.h>

// ---------------------------------------------------------------------------
// Causal MHA for MI455X (gfx1250), f16 WMMA w/ f32 accumulation.
//   1) cvt x -> f16; transpose+cvt Wq/Wk/Wv/Wp -> f16 [N][K]
//   2) WMMA GEMM (32x64 tile/wave, reg double-buffered):
//        Q,K -> [B,H,T,D] f16 ; V -> [B,H,D,T] f16 (transposed epilogue)
//   3) flash attention: block = (b,h) x 8 q-tiles; K/V tiles staged into LDS
//      with global_load_async_to_lds_b128 (ping-pong, s_wait_asynccnt),
//      online softmax, P relayout through LDS, PV via WMMA
//   4) WMMA GEMM: y_h @ Wp_t + bp -> d_out (f32)
// ---------------------------------------------------------------------------

typedef __attribute__((ext_vector_type(16))) _Float16 v16h;
typedef __attribute__((ext_vector_type(8)))  _Float16 v8h;
typedef __attribute__((ext_vector_type(8)))  float    v8f;

#define BB 4
#define TT 2048
#define CC 1024
#define HH 16
#define DD 64
#define BT (BB * TT)   // 8192 rows

__device__ __forceinline__ v8f wmma32(v16h a, v16h b, v8f c) {
    return __builtin_amdgcn_wmma_f32_16x16x32_f16(false, a, false, b,
                                                  (short)0, c, false, false);
}

// 16-bit A/B fragment: two contiguous 8-half (16B) chunks at p and p+16 halfs
// (caller applies the lane-half K-offset). Works for global or LDS pointers.
__device__ __forceinline__ v16h ld_frag(const _Float16* p) {
    v8h lo = *(const v8h*)(p);
    v8h hi = *(const v8h*)(p + 16);
    return __builtin_shufflevector(lo, hi, 0, 1, 2, 3, 4, 5, 6, 7,
                                   8, 9, 10, 11, 12, 13, 14, 15);
}

// ---------------------------------------------------------------------------
__global__ __launch_bounds__(256) void k_cvt(const float* __restrict__ in,
                                             _Float16* __restrict__ out, int n) {
    int i = blockIdx.x * 256 + threadIdx.x;
    if (i < n) out[i] = (_Float16)in[i];
}

// W[k][n] (f32) -> Wt[n][k] (f16): B-fragments become contiguous b128 loads.
__global__ __launch_bounds__(256) void k_cvt_t(const float* __restrict__ W,
                                               _Float16* __restrict__ Wt) {
    int i = blockIdx.x * 256 + threadIdx.x;
    int n = i >> 10;
    int k = i & (CC - 1);
    Wt[i] = (_Float16)W[k * CC + n];
}

// ---------------------------------------------------------------------------
// GEMM: OUT[m,n] = sum_k X[m,k] * Wt[n,k] + bias[n]
//   mode 0: f16 out [B,H,T,D] (Q,K); mode 1: f16 out [B,H,D,T] (V);
//   mode 2: f32 out [B,T,C] (final projection)
// One wave: 32x64 tile, 8 WMMAs per 32-K step, register double-buffering.
__global__ __launch_bounds__(256) void k_gemm(const _Float16* __restrict__ X,
                                              const _Float16* __restrict__ Wt,
                                              const float* __restrict__ bias,
                                              _Float16* __restrict__ out16,
                                              float* __restrict__ out32,
                                              int mode) {
    int wave = (blockIdx.x * 256 + (int)threadIdx.x) >> 5;
    int lane = threadIdx.x & 31;
    int mt = wave >> 4;        // 256 M-tiles of 32 rows
    int ng = wave & 15;        // 16 N-groups of 64 cols
    if (mt >= BT / 32) return;

    int m0 = mt * 32, n0 = ng * 64;
    int nlane = lane & 15;
    int hs8 = (lane >> 4) << 3;     // 0/8: K-chunk offset AND C-row offset

    const _Float16* xr0 = X + (m0 + nlane) * CC + hs8;
    const _Float16* xr1 = X + (m0 + 16 + nlane) * CC + hs8;
    const _Float16* w0  = Wt + (n0 + nlane) * CC + hs8;

    v8f acc[2][4] = {};
    // prologue: fragments for kk = 0
    v16h a0 = ld_frag(xr0), a1 = ld_frag(xr1);
    v16h b[4];
#pragma unroll
    for (int j = 0; j < 4; j++) b[j] = ld_frag(w0 + j * 16 * CC);

#pragma unroll 2
    for (int kk = 0; kk < CC - 32; kk += 32) {
        int kn = kk + 32;
        // prefetch next K-step's fragments before consuming current ones
        v16h a0n = ld_frag(xr0 + kn);
        v16h a1n = ld_frag(xr1 + kn);
        v16h bn[4];
#pragma unroll
        for (int j = 0; j < 4; j++) bn[j] = ld_frag(w0 + j * 16 * CC + kn);
#pragma unroll
        for (int j = 0; j < 4; j++) {
            acc[0][j] = wmma32(a0, b[j], acc[0][j]);
            acc[1][j] = wmma32(a1, b[j], acc[1][j]);
        }
        a0 = a0n; a1 = a1n;
#pragma unroll
        for (int j = 0; j < 4; j++) b[j] = bn[j];
    }
#pragma unroll
    for (int j = 0; j < 4; j++) {       // tail K-step
        acc[0][j] = wmma32(a0, b[j], acc[0][j]);
        acc[1][j] = wmma32(a1, b[j], acc[1][j]);
    }

    float bj[4];
#pragma unroll
    for (int j = 0; j < 4; j++) bj[j] = bias[n0 + j * 16 + nlane];

    int h = n0 >> 6;
#pragma unroll
    for (int mi = 0; mi < 2; mi++) {
#pragma unroll
        for (int r = 0; r < 8; r++) {
            int m = m0 + mi * 16 + r + hs8;
            int bidx = m >> 11;            // /T
            int t = m & (TT - 1);          // %T
#pragma unroll
            for (int j = 0; j < 4; j++) {
                float v = acc[mi][j][r] + bj[j];
                int d = j * 16 + nlane;
                if (mode == 2) {
                    out32[m * CC + n0 + d] = v;
                } else if (mode == 0) {
                    out16[((bidx * HH + h) * TT + t) * DD + d] = (_Float16)v;
                } else {
                    out16[((bidx * HH + h) * DD + d) * TT + t] = (_Float16)v;
                }
            }
        }
    }
}

// ---------------------------------------------------------------------------
// Cooperative async stage of one 32-key step: K tile (32x64 halfs) and
// V tile (64x32 halfs) into LDS. 256 threads x 16B each per tile
// => exactly 2 global_load_async_to_lds_b128 per thread (tracked by ASYNCcnt).
__device__ __forceinline__ void stage_tiles(const _Float16* Kb, const _Float16* Vb,
                                            _Float16* kl, _Float16* vl,
                                            int k0, int tid) {
    {   // K: 32 rows x 128B
        int row = tid >> 3, chunk = (tid & 7) << 3;          // chunk in halfs
        const _Float16* g = Kb + (k0 + row) * DD + chunk;
        unsigned l = (unsigned)(uintptr_t)(kl + row * DD + chunk);
        asm volatile("global_load_async_to_lds_b128 %0, %1, off"
                     :: "v"(l), "v"(g) : "memory");
    }
    {   // V ([d][t] layout): 64 rows x 64B
        int row = tid >> 2, chunk = (tid & 3) << 3;
        const _Float16* g = Vb + (size_t)row * TT + k0 + chunk;
        unsigned l = (unsigned)(uintptr_t)(vl + row * 32 + chunk);
        asm volatile("global_load_async_to_lds_b128 %0, %1, off"
                     :: "v"(l), "v"(g) : "memory");
    }
}

// Flash attention. Block = (b,h) x 8 consecutive 16-row q-tiles (1 per wave).
// K/V tiles double-buffered in LDS via async loads shared by all 8 waves.
__global__ __launch_bounds__(256) void k_attn(const _Float16* __restrict__ Q,
                                              const _Float16* __restrict__ K,
                                              const _Float16* __restrict__ Vt,
                                              _Float16* __restrict__ Y) {
    __shared__ __attribute__((aligned(16))) _Float16 Kl[2][32 * DD];   // 2x4KB
    __shared__ __attribute__((aligned(16))) _Float16 Vl[2][DD * 32];   // 2x4KB
    __shared__ __attribute__((aligned(16))) _Float16 Pst[8][16 * 32];  // 8x1KB

    int tid = threadIdx.x;
    int lane = tid & 31;
    int wib = tid >> 5;
    int bh = blockIdx.x >> 4;                 // b*H + h
    int qtbase = (blockIdx.x & 15) * 8;
    int qt = qtbase + wib;                    // this wave's 16-row q-tile

    int nlane = lane & 15;
    int hs8 = (lane >> 4) << 3;

    const _Float16* Qb = Q + (size_t)bh * TT * DD;
    const _Float16* Kb = K + (size_t)bh * TT * DD;
    const _Float16* Vb = Vt + (size_t)bh * DD * TT;

    const _Float16* qp = Qb + (qt * 16 + nlane) * DD;
    v16h qa0 = ld_frag(qp + hs8);
    v16h qa1 = ld_frag(qp + 32 + hs8);

    v8f o[4] = {};
    float mrow[8], lrow[8];
#pragma unroll
    for (int r = 0; r < 8; r++) { mrow[r] = -1e30f; lrow[r] = 0.0f; }

    _Float16* P = &Pst[wib][0];
    int mySteps = ((qt * 16 + 15) >> 5) + 1;
    int blkSteps = (((qtbase + 7) * 16 + 15) >> 5) + 1;

    stage_tiles(Kb, Vb, Kl[0], Vl[0], 0, tid);

    for (int s = 0; s < blkSteps; s++) {
        int k0 = s * 32;
        __syncthreads();   // all readers of buffer (s+1)&1 (step s-1) are done
        if (s + 1 < blkSteps) {
            stage_tiles(Kb, Vb, Kl[(s + 1) & 1], Vl[(s + 1) & 1], k0 + 32, tid);
            asm volatile("s_wait_asynccnt 2" ::: "memory");   // step-s loads landed
        } else {
            asm volatile("s_wait_asynccnt 0" ::: "memory");
        }
        __syncthreads();   // every thread's step-s tile data visible in LDS

        if (s < mySteps) {
            const _Float16* kl = Kl[s & 1];
            const _Float16* vl = Vl[s & 1];

            // scores: two 16x16 tiles over keys [k0, k0+32), frags from LDS
            v8f s0 = {}, s1 = {};
            s0 = wmma32(qa0, ld_frag(kl + nlane * DD + hs8), s0);
            s0 = wmma32(qa1, ld_frag(kl + nlane * DD + 32 + hs8), s0);
            s1 = wmma32(qa0, ld_frag(kl + (16 + nlane) * DD + hs8), s1);
            s1 = wmma32(qa1, ld_frag(kl + (16 + nlane) * DD + 32 + hs8), s1);

            const float scale = 0.125f;   // D^-0.5
#pragma unroll
            for (int r = 0; r < 8; r++) {
                int qrow = qt * 16 + r + hs8;
                float a = s0[r] * scale;
                float b = s1[r] * scale;
                if (k0 + nlane > qrow) a = -1e30f;
                if (k0 + 16 + nlane > qrow) b = -1e30f;
                float mx = fmaxf(a, b);
#pragma unroll
                for (int off = 1; off < 16; off <<= 1)
                    mx = fmaxf(mx, __shfl_xor(mx, off, 32));
                float mnew = fmaxf(mrow[r], mx);
                float corr = __expf(mrow[r] - mnew);
                float pa = __expf(a - mnew);
                float pb = __expf(b - mnew);
                float ls = pa + pb;
#pragma unroll
                for (int off = 1; off < 16; off <<= 1)
                    ls += __shfl_xor(ls, off, 32);
                lrow[r] = lrow[r] * corr + ls;
                mrow[r] = mnew;
#pragma unroll
                for (int j = 0; j < 4; j++) o[j][r] *= corr;
                int prow = r + hs8;
                P[prow * 32 + nlane] = (_Float16)pa;
                P[prow * 32 + 16 + nlane] = (_Float16)pb;
            }
            asm volatile("s_wait_dscnt 0" ::: "memory");  // own-wave P staging

            // O += P(16x32) @ V(32x64); V rows are [d][key] in LDS
            v16h pA = ld_frag(P + nlane * 32 + hs8);
#pragma unroll
            for (int j = 0; j < 4; j++) {
                v16h vb = ld_frag(vl + (j * 16 + nlane) * 32 + hs8);
                o[j] = wmma32(pA, vb, o[j]);
            }
        }
    }

    int h = bh & (HH - 1);
    int b = bh >> 4;
#pragma unroll
    for (int r = 0; r < 8; r++) {
        float inv = 1.0f / lrow[r];
        int t = qt * 16 + r + hs8;
        size_t base = ((size_t)b * TT + t) * CC + h * DD;
#pragma unroll
        for (int j = 0; j < 4; j++)
            Y[base + j * 16 + nlane] = (_Float16)(o[j][r] * inv);
    }
}

// ---------------------------------------------------------------------------
extern "C" void kernel_launch(void* const* d_in, const int* in_sizes, int n_in,
                              void* d_out, int out_size, void* d_ws, size_t ws_size,
                              hipStream_t stream) {
    const float* x  = (const float*)d_in[0];
    const float* Wq = (const float*)d_in[1];
    const float* bq = (const float*)d_in[2];
    const float* Wk = (const float*)d_in[3];
    const float* bk = (const float*)d_in[4];
    const float* Wv = (const float*)d_in[5];
    const float* bv = (const float*)d_in[6];
    const float* Wp = (const float*)d_in[7];
    const float* bp = (const float*)d_in[8];

    char* ws = (char*)d_ws;
    const size_t MB = (size_t)1 << 20;
    _Float16* xh  = (_Float16*)(ws);              // 16 MB  (B*T*C f16)
    _Float16* wqt = (_Float16*)(ws + 16 * MB);    //  2 MB each (C*C f16)
    _Float16* wkt = (_Float16*)(ws + 18 * MB);
    _Float16* wvt = (_Float16*)(ws + 20 * MB);
    _Float16* wpt = (_Float16*)(ws + 22 * MB);
    _Float16* Qh  = (_Float16*)(ws + 24 * MB);    // 16 MB [B,H,T,D]
    _Float16* Kh  = (_Float16*)(ws + 40 * MB);    // 16 MB [B,H,T,D]
    _Float16* Vth = (_Float16*)(ws + 56 * MB);    // 16 MB [B,H,D,T]
    _Float16* Yh  = (_Float16*)(ws + 72 * MB);    // 16 MB [B,T,C]

    int nx = BB * TT * CC;
    k_cvt<<<nx / 256, 256, 0, stream>>>(x, xh, nx);
    int nw = CC * CC;
    k_cvt_t<<<nw / 256, 256, 0, stream>>>(Wq, wqt);
    k_cvt_t<<<nw / 256, 256, 0, stream>>>(Wk, wkt);
    k_cvt_t<<<nw / 256, 256, 0, stream>>>(Wv, wvt);
    k_cvt_t<<<nw / 256, 256, 0, stream>>>(Wp, wpt);

    // GEMMs: 4096 waves (256 Mtiles x 16 Ngroups) = 512 blocks x 8 waves
    k_gemm<<<512, 256, 0, stream>>>(xh, wqt, bq, Qh, nullptr, 0);
    k_gemm<<<512, 256, 0, stream>>>(xh, wkt, bk, Kh, nullptr, 0);
    k_gemm<<<512, 256, 0, stream>>>(xh, wvt, bv, Vth, nullptr, 1);
    // attention: 64 (b,h) x 16 q-tile-groups = 1024 blocks
    k_attn<<<1024, 256, 0, stream>>>(Qh, Kh, Vth, Yh);
    k_gemm<<<512, 256, 0, stream>>>(Yh, wpt, bp, nullptr, (float*)d_out, 2);
}
